// ncc3d_14972255993950
// MI455X (gfx1250) — matrix operand order repaired and checked by reading, not compiled, not run on hardware
//
#include <hip/hip_runtime.h>

typedef __attribute__((ext_vector_type(16))) _Float16 v16h;
typedef __attribute__((ext_vector_type(8)))  float    v8f;
typedef __attribute__((ext_vector_type(4)))  unsigned u32x4;
typedef __attribute__((ext_vector_type(8)))  int      i32x8;
typedef __attribute__((ext_vector_type(4)))  int      i32x4;

#define NCC_EPS 1e-5f

#if defined(__gfx1250__) && __has_builtin(__builtin_amdgcn_tensor_load_to_lds) && \
    __has_builtin(__builtin_amdgcn_s_wait_tensorcnt)
#define USE_TDM 1
#else
#define USE_TDM 0
#endif

__device__ __forceinline__ float fast_sqrtf(float x) {
#if defined(__gfx1250__) && __has_builtin(__builtin_amdgcn_sqrtf)
    return __builtin_amdgcn_sqrtf(x);
#else
    return sqrtf(x);
#endif
}
__device__ __forceinline__ float fast_rcpf(float x) {
#if defined(__gfx1250__) && __has_builtin(__builtin_amdgcn_rcpf)
    return __builtin_amdgcn_rcpf(x);
#else
    return 1.0f / x;
#endif
}
__device__ __forceinline__ int mod12(int v) { return v >= 12 ? v - 12 : v; }

// ---------------------------------------------------------------------------
// f32 -> f16 conversion pass (enables TDM byte-copy staging of x planes)
// ---------------------------------------------------------------------------
__global__ __launch_bounds__(256)
void ncc_cvt(const float* __restrict__ x, _Float16* __restrict__ xh)
{
    const size_t i = ((size_t)blockIdx.x * 256 + threadIdx.x) * 4;
    const float4 v = *(const float4*)(x + i);
    union { _Float16 h[4]; uint2 u; } p;
    p.h[0] = (_Float16)v.x; p.h[1] = (_Float16)v.y;
    p.h[2] = (_Float16)v.z; p.h[3] = (_Float16)v.w;
    *(uint2*)(xh + i) = p.u;
}

// ---------------------------------------------------------------------------
// Prep: zero-mean template, z_norm, 64 Toeplitz B tiles per batch in the
// contiguous-k wave32 B-matrix register layout (A loads in ncc_main match):
//   lane L holds column N = L%16;  K = 2*v + h + 16*(L/16)
//   B[K][N] = zc[kd][kh][K-N] if 0 <= K-N < 8 else 0
// ---------------------------------------------------------------------------
__global__ __launch_bounds__(256)
void ncc_prep(const float* __restrict__ z, unsigned* __restrict__ wsB,
              float* __restrict__ znorm)
{
    const int b = blockIdx.x;
    const int t = threadIdx.x;
    __shared__ float zs[512];
    __shared__ float red[256];

    const float* zb = z + b * 512;
    zs[t]       = zb[t];
    zs[t + 256] = zb[t + 256];
    __syncthreads();
    red[t] = zs[t] + zs[t + 256];
    __syncthreads();
    for (int off = 128; off > 0; off >>= 1) {
        if (t < off) red[t] += red[t + off];
        __syncthreads();
    }
    const float mean = red[0] * (1.0f / 512.0f);
    __syncthreads();
    const float c0 = zs[t] - mean;
    const float c1 = zs[t + 256] - mean;
    red[t]      = c0 * c0 + c1 * c1;
    zs[t]       = c0;
    zs[t + 256] = c1;
    __syncthreads();
    for (int off = 128; off > 0; off >>= 1) {
        if (t < off) red[t] += red[t + off];
        __syncthreads();
    }
    if (t == 0) znorm[b] = sqrtf(red[0]);

    unsigned* Bb = wsB + b * 16384;
    for (int idx = t; idx < 16384; idx += 256) {
        const int v    = idx & 7;
        const int lane = (idx >> 3) & 31;
        const int kdkh = idx >> 8;
        const int N    = lane & 15;
        const int K0   = 2 * v + ((lane >> 4) << 4);
        union { _Float16 h[2]; unsigned u; } pk;
#pragma unroll
        for (int j = 0; j < 2; ++j) {
            const int kw = (K0 + j) - N;
            pk.h[j] = (kw >= 0 && kw < 8) ? (_Float16)zs[kdkh * 8 + kw]
                                          : (_Float16)0.0f;
        }
        Bb[idx] = pk.u;
    }
}

// ---------------------------------------------------------------------------
// Main: one wave32 per workgroup, grid (64, 8, 8).
// Processes TWO od planes per iteration with two independent WMMA
// accumulator chains; each staged A tile feeds both (B[t] / B[t-1]).
// 12-deep LDS plane ring, TDM prefetch two planes ahead, rolling separable
// window sums for the NCC denominator.
// ---------------------------------------------------------------------------
__global__ __launch_bounds__(32)
void ncc_main(const _Float16* __restrict__ xh, const unsigned* __restrict__ wsB,
              const float* __restrict__ znorm, float* __restrict__ out,
              float* __restrict__ dump)
{
    const int lane = threadIdx.x;          // 0..31
    const int b    = blockIdx.z;
    const int od0  = blockIdx.y * 16;
    const int oh0  = (blockIdx.x >> 3) * 16;
    const int ow0  = (blockIdx.x & 7) * 16;
    const int odEnd = (od0 + 16 < 121) ? od0 + 16 : 121;

    __shared__ __align__(16) _Float16 slab[12][23][40]; // 64B row + 16B pad
    __shared__ float p2x[12][16][16];
    __shared__ float p2q[12][16][16];
    __shared__ float p1x[23][16];
    __shared__ float p1q[23][16];

    const _Float16* xb = xh + ((size_t)b << 21);        // 128^3 per batch
    const unsigned* Bb = wsB + b * 16384;
    const int g = lane >> 4;
    const int N = lane & 15;

    // ---- stage plane d (23 x 32 halves) into ring slot ----
    auto plane_issue = [&](int d, int slot) {
        if (d > 127) d = 127;                            // dead-tail clamp
#if USE_TDM
        const unsigned lds_off =
            (unsigned)(size_t)&slab[slot][0][0];         // LDS byte offset
        const unsigned long long ga = (unsigned long long)(size_t)
            (xb + (((size_t)d) << 14) + oh0 * 128 + ow0);
        u32x4 g0;
        g0[0] = 1u;                                      // count=1, user mode
        g0[1] = lds_off;                                 // lds_addr
        g0[2] = (unsigned)ga;                            // global_addr[31:0]
        g0[3] = (unsigned)((ga >> 32) & 0x01FFFFFFu) | (2u << 30); // type=2
        i32x8 g1;
        g1[0] = (int)((1u << 16) |                       // data_size = 2B
                      (1u << 20) |                       // pad_enable
                      (3u << 22) |                       // pad_interval: 16 DW
                      (3u << 25));                       // pad_amount:   4 DW
        g1[1] = (int)((unsigned)(128 - ow0) << 16);      // tensor_dim0 (OOB->0)
        g1[2] = (int)((unsigned)(128 - oh0) << 16);      // tensor_dim1 (OOB->0)
        g1[3] = (int)(32u << 16);                        // tile_dim0 = 32
        g1[4] = 23;                                      // tile_dim1 = 23
        g1[5] = 128;                                     // tensor_dim0_stride
        g1[6] = 0;
        g1[7] = 0;
        const i32x4 gz4 = {0, 0, 0, 0};
        const i32x8 gz8 = {0, 0, 0, 0, 0, 0, 0, 0};
        __builtin_amdgcn_tensor_load_to_lds(g0, g1, gz4, gz4, gz8, 0);
#else
        __builtin_prefetch(xb + (((size_t)d) << 14), 0, 1);
        for (int i = lane; i < 23 * 32; i += 32) {
            const int h = i >> 5, w = i & 31;
            const int gh = oh0 + h, gw = ow0 + w;
            _Float16 v = (_Float16)0.0f;
            if (gh < 128 && gw < 128)
                v = xb[(((size_t)d) << 14) + (gh << 7) + gw];
            slab[slot][h][w] = v;
        }
#endif
    };
    auto plane_wait = [&]() {
#if USE_TDM
        __builtin_amdgcn_s_wait_tensorcnt(0);
#endif
        __syncthreads();
    };
    // ---- separable W/H window sums of a resident plane (x and x^2) ----
    auto plane_stats = [&](int slot) {
        for (int i = lane; i < 23 * 16; i += 32) {
            const int h = i >> 4, w = i & 15;
            float s = 0.f, sq = 0.f;
#pragma unroll
            for (int kw = 0; kw < 8; ++kw) {
                const float v = (float)slab[slot][h][w + kw];
                s += v; sq += v * v;
            }
            p1x[h][w] = s; p1q[h][w] = sq;
        }
        __syncthreads();
        for (int i = lane; i < 256; i += 32) {
            const int oh = i >> 4, ow = i & 15;
            float s = 0.f, sq = 0.f;
#pragma unroll
            for (int kh = 0; kh < 8; ++kh) {
                s += p1x[oh + kh][ow]; sq += p1q[oh + kh][ow];
            }
            p2x[slot][oh][ow] = s; p2q[slot][oh][ow] = sq;
        }
        __syncthreads();
    };
    auto wmma_step = [&](v8f& acc, int slot, int kd, int kh) {
        v16h a, bm;
        const _Float16* pa = &slab[slot][N + kh][g * 16];
        ((int4*)&a)[0] = *(const int4*)pa;        // K = 16g .. 16g+7
        ((int4*)&a)[1] = *(const int4*)(pa + 8);  // K = 16g+8 .. 16g+15
        const uint4* pb =
            (const uint4*)(Bb + (((kd * 8 + kh) * 32 + lane) << 3));
        ((uint4*)&bm)[0] = pb[0];
        ((uint4*)&bm)[1] = pb[1];
        acc = __builtin_amdgcn_wmma_f32_16x16x32_f16(
            false, a, false, bm, (short)0, acc, false, false);
    };

    // ---- prologue: planes od0 .. od0+8 ----
    int rs = od0 % 12;
    for (int t = 0; t < 9; ++t) plane_issue(od0 + t, mod12(rs + t));
    plane_wait();
    for (int t = 0; t < 9; ++t) plane_stats(mod12(rs + t));

    float s1[8], s2[8];
#pragma unroll
    for (int v = 0; v < 8; ++v) {
        const int M = v + (g << 3);
        float a = 0.f, q = 0.f;
#pragma unroll
        for (int t = 0; t < 8; ++t) {
            const int s = mod12(rs + t);
            a += p2x[s][M][N]; q += p2q[s][M][N];
        }
        s1[v] = a; s2[v] = q;
    }

    const float  zn   = znorm[b];
    const size_t outB = (size_t)b * (121 * 121 * 121);

    for (int od = od0; od < odEnd; od += 2) {
        const bool issue_next = (od + 2 < odEnd);
        if (issue_next) {                   // async copies overlap the WMMAs
            plane_issue(od + 9,  mod12(rs + 9));
            plane_issue(od + 10, mod12(rs + 10));
        }

        v8f acc0 = {}, acc1 = {};
        {   // t = 0: plane od feeds only acc0 (kd = 0)
#pragma unroll
            for (int kh = 0; kh < 8; ++kh) wmma_step(acc0, rs, 0, kh);
        }
#pragma unroll 1
        for (int t = 1; t < 8; ++t) {       // shared A -> both accumulators
            const int slot = mod12(rs + t);
#pragma unroll
            for (int kh = 0; kh < 8; ++kh) {
                wmma_step(acc0, slot, t, kh);
                wmma_step(acc1, slot, t - 1, kh);
            }
        }
        {   // t = 8: plane od+8 feeds only acc1 (kd = 7)
            const int slot = mod12(rs + 8);
#pragma unroll
            for (int kh = 0; kh < 8; ++kh) wmma_step(acc1, slot, 7, kh);
        }

        // ---- epilogue od (acc0, window sums s1/s2) ----
#pragma unroll
        for (int v = 0; v < 8; ++v) {
            const int M  = v + (g << 3);
            const int oh = oh0 + M, ow = ow0 + N;
            const float sx = s1[v];
            float var = s2[v] - sx * sx * (1.0f / 512.0f);
            var = fmaxf(var, 0.0f);
            const float inv =
                fast_rcpf(fast_sqrtf(var + 1e-12f) * zn + NCC_EPS);
            float* po = out + (outB + (size_t)od * (121 * 121) + oh * 121 + ow);
            float* p  = (oh < 121 && ow < 121) ? po : (dump + lane);
            *p = acc0[v] * inv;
        }

        // ---- shift sums to od+1 and store acc1 ----
        const int slotA = rs;               // plane od
        const int slotB = mod12(rs + 8);    // plane od+8
        const bool odv  = (od + 1 < odEnd);
#pragma unroll
        for (int v = 0; v < 8; ++v) {
            const int M  = v + (g << 3);
            s1[v] += p2x[slotB][M][N] - p2x[slotA][M][N];
            s2[v] += p2q[slotB][M][N] - p2q[slotA][M][N];
            const int oh = oh0 + M, ow = ow0 + N;
            const float sx = s1[v];
            float var = s2[v] - sx * sx * (1.0f / 512.0f);
            var = fmaxf(var, 0.0f);
            const float inv =
                fast_rcpf(fast_sqrtf(var + 1e-12f) * zn + NCC_EPS);
            float* po = out + (outB + (size_t)(od + 1) * (121 * 121) +
                               oh * 121 + ow);
            float* p  = (odv && oh < 121 && ow < 121) ? po : (dump + lane);
            *p = acc1[v] * inv;
        }

        // ---- retire TDMs, stats for planes od+9 / od+10, sums -> od+2 ----
        if (issue_next) {
            plane_wait();
            plane_stats(mod12(rs + 9));
            plane_stats(mod12(rs + 10));
            const int slotC = mod12(rs + 1); // plane od+1
            const int slotD = mod12(rs + 9); // plane od+9
#pragma unroll
            for (int v = 0; v < 8; ++v) {
                const int M = v + (g << 3);
                s1[v] += p2x[slotD][M][N] - p2x[slotC][M][N];
                s2[v] += p2q[slotD][M][N] - p2q[slotC][M][N];
            }
        }
        rs += 2; if (rs >= 12) rs -= 12;
    }
}

// ---------------------------------------------------------------------------
extern "C" void kernel_launch(void* const* d_in, const int* in_sizes, int n_in,
                              void* d_out, int out_size, void* d_ws, size_t ws_size,
                              hipStream_t stream)
{
    const float* z = (const float*)d_in[0];   // [8,1,8,8,8]
    const float* x = (const float*)d_in[1];   // [8,1,128,128,128]
    float*       o = (float*)d_out;           // [8,1,121,121,121]

    // ws layout: [0,512K) Toeplitz B | 512K: znorm(32B) | +64: dump(128B)
    //            [1M, 1M+32M): f16 copy of x
    unsigned* wsB   = (unsigned*)d_ws;
    float*    znorm = (float*)((char*)d_ws + 8 * 16384 * 4);
    float*    dump  = (float*)((char*)d_ws + 8 * 16384 * 4 + 64);
    _Float16* xh    = (_Float16*)((char*)d_ws + (1u << 20));

    ncc_cvt <<<16384, 256, 0, stream>>>(x, xh);
    ncc_prep<<<8,     256, 0, stream>>>(z, wsB, znorm);

    dim3 grid(64, 8, 8);   // (oh_tile*8+ow_tile, od_chunk, batch)
    ncc_main<<<grid, 32, 0, stream>>>(xh, wsB, znorm, o, dump);
}
